// BGRL_35218731827951
// MI455X (gfx1250) — matrix operand order repaired and checked by reading, not compiled, not run on hardware
//
#include <hip/hip_runtime.h>

// ---------------------------------------------------------------------------
// BGRL step for MI455X (gfx1250): bf16 WMMA GEMMs + atomic edge aggregation.
// ---------------------------------------------------------------------------

#define DN 128          // feature dim
#define NROW 50000      // nodes
#define NEDGE 800000    // edges

typedef __attribute__((ext_vector_type(16))) __bf16 v16bf;
typedef __attribute__((ext_vector_type(8)))  float  v8f;

union V16 {
    v16bf v;
    uint4 q[2];
    unsigned short s[16];
};

static __device__ __forceinline__ unsigned short f2bf(float f) {
    unsigned u = __builtin_bit_cast(unsigned, f);
    u += 0x7FFFu + ((u >> 16) & 1u);          // round-to-nearest-even
    return (unsigned short)(u >> 16);
}
static __device__ __forceinline__ float bf2f(unsigned short h) {
    unsigned u = ((unsigned)h) << 16;
    return __builtin_bit_cast(float, u);
}

// ---------------------------------------------------------------------------
// 1) Elementwise prep: xb = bf16(x), x2b = bf16(x+perb); zero accumulators.
// ---------------------------------------------------------------------------
__global__ void __launch_bounds__(256)
prep_main(const float* __restrict__ x, const float* __restrict__ perb,
          unsigned short* __restrict__ xb, unsigned short* __restrict__ x2b,
          float* __restrict__ A0, float* __restrict__ A1,
          float* __restrict__ A2, float* __restrict__ A3,
          float* __restrict__ stats, float* __restrict__ lossOut, int nd)
{
    int i = blockIdx.x * blockDim.x + threadIdx.x;
    if (i >= nd) return;
    float xv  = x[i];
    float x2v = xv + perb[i];
    xb[i]  = f2bf(xv);
    x2b[i] = f2bf(x2v);
    A0[i] = 0.f; A1[i] = 0.f; A2[i] = 0.f; A3[i] = 0.f;
    if (i < 512) stats[i] = 0.f;
    if (i == 0)  *lossOut = 0.f;
}

// ---------------------------------------------------------------------------
// 2) Weight prep: bf16 + transpose (WT[n*128+k] = W[k*128+n]) for 4 weights.
// ---------------------------------------------------------------------------
__global__ void __launch_bounds__(256)
prep_w(const float* __restrict__ W0, const float* __restrict__ W1,
       const float* __restrict__ W2, const float* __restrict__ W3,
       unsigned short* __restrict__ T0, unsigned short* __restrict__ T1,
       unsigned short* __restrict__ T2, unsigned short* __restrict__ T3)
{
    int i = blockIdx.x * blockDim.x + threadIdx.x;  // i = k*128 + n
    if (i >= DN * DN) return;
    int k = i >> 7, n = i & 127;
    int o = n * DN + k;
    T0[o] = f2bf(W0[i]);
    T1[o] = f2bf(W1[i]);
    T2[o] = f2bf(W2[i]);
    T3[o] = f2bf(W3[i]);
}

// ---------------------------------------------------------------------------
// 3) bf16 WMMA GEMM: out[nrows x 128] = A[nrows x 128] @ W[128 x 128] (+bias)
//    WT is the transposed bf16 weight (column-major of W).
//    Block = 256 threads = 8 waves; wave owns a 16-row strip x 128 cols.
//    Weight staged in LDS (32 KB); K-loop = 4 x wmma_f32_16x16x32_bf16.
//    Templated on output type / bias so the epilogue is branch-free and all
//    stores use compile-time immediate offsets.
// ---------------------------------------------------------------------------
template <bool OUT_BF16, bool HAS_BIAS>
__global__ void __launch_bounds__(256)
gemm_wmma(const unsigned short* __restrict__ A, const unsigned short* __restrict__ WT,
          const float* __restrict__ bias,
          unsigned short* __restrict__ outb, float* __restrict__ outf, int nrows)
{
    __shared__ __attribute__((aligned(16))) unsigned short ldsW[DN * DN];

    int tid = threadIdx.x;
    {   // cooperative stage: 16384 ushorts = 2048 uint4
        const uint4* src = (const uint4*)WT;
        uint4*       dst = (uint4*)ldsW;
#pragma unroll
        for (int i = 0; i < 8; ++i) dst[tid + 256 * i] = src[tid + 256 * i];
    }
    __syncthreads();

    int wave = tid >> 5, lane = tid & 31;
    int m0 = (blockIdx.x * 8 + wave) * 16;
    if (m0 >= nrows) return;   // nrows % 16 == 0, so full tiles only

    // A fragment (16x32 bf16, ISA 7.12.2): lane<16 -> row=lane, K={0..7,16..23};
    // lane>=16 -> K shifted by 8. Two 16B chunks per fragment.
    int mrow = lane & 15;
    int koff = (lane >> 4) * 8;
    v16bf afrag[4];
    const unsigned short* arow = A + (size_t)(m0 + mrow) * DN;
#pragma unroll
    for (int kk = 0; kk < 4; ++kk) {
        const unsigned short* ap = arow + kk * 32 + koff;
        V16 t;
        t.q[0] = *(const uint4*)ap;
        t.q[1] = *(const uint4*)(ap + 16);
        afrag[kk] = t.v;
    }

    // B fragment (32x16 bf16): lane holds column n=lane&15; K={0..7,16..23},
    // lane>=16 -> K shifted by 8 (matches A) -> two uint4 LDS reads.
    int bk = (lane >> 4) * 8;
    int rbase = m0 + (lane >> 4) * 8;     // first output row owned by this lane
#pragma unroll
    for (int nt = 0; nt < 8; ++nt) {
        int n0  = nt * 16;
        int col = n0 + (lane & 15);
        v8f c = {0.f, 0.f, 0.f, 0.f, 0.f, 0.f, 0.f, 0.f};
#pragma unroll
        for (int kk = 0; kk < 4; ++kk) {
            const unsigned short* bp = &ldsW[col * DN + kk * 32 + bk];
            V16 b;
            b.q[0] = *(const uint4*)bp;
            b.q[1] = *(const uint4*)(bp + 16);
            c = __builtin_amdgcn_wmma_f32_16x16x32_bf16(
                    false, afrag[kk], false, b.v, (short)0, c, false, false);
        }
        // C/D layout: VGPR r, lanes0-15 -> M=r, lanes16-31 -> M=r+8; N=lane&15
        float bv = HAS_BIAS ? bias[col] : 0.0f;
        size_t base = (size_t)rbase * DN + col;
        if constexpr (OUT_BF16) {
            unsigned short* p = outb + base;
#pragma unroll
            for (int r = 0; r < 8; ++r) p[r * DN] = f2bf(c[r] + bv);
        } else {
            float* p = outf + base;
#pragma unroll
            for (int r = 0; r < 8; ++r) p[r * DN] = c[r] + bv;
        }
    }
}

// ---------------------------------------------------------------------------
// 4) Edge aggregation: wave per edge, lane covers 4 features.
//    Oa[row] += Ha[col]*w ; Ob[row] += Hb[col]*w (fp32 global atomics).
// ---------------------------------------------------------------------------
__global__ void __launch_bounds__(256)
agg2_kernel(const unsigned short* __restrict__ Ha, const unsigned short* __restrict__ Hb,
            const int* __restrict__ er, const int* __restrict__ ec,
            const float* __restrict__ ew,
            float* __restrict__ Oa, float* __restrict__ Ob, int nE)
{
    int gid  = blockIdx.x * blockDim.x + threadIdx.x;
    int e    = gid >> 5;
    int lane = threadIdx.x & 31;
    if (e >= nE) return;
    int   r = er[e], c = ec[e];
    float w = ew[e];
    int co = lane * 4;
    uint2 pa = *(const uint2*)(Ha + (size_t)c * DN + co);
    uint2 pb = *(const uint2*)(Hb + (size_t)c * DN + co);
    float* oa = Oa + (size_t)r * DN + co;
    float* ob = Ob + (size_t)r * DN + co;
    unsigned sa[2] = {pa.x, pa.y};
    unsigned sb[2] = {pb.x, pb.y};
#pragma unroll
    for (int j = 0; j < 4; ++j) {
        unsigned ua = (sa[j >> 1] >> ((j & 1) * 16)) & 0xFFFFu;
        unsigned ub = (sb[j >> 1] >> ((j & 1) * 16)) & 0xFFFFu;
        atomicAdd(oa + j, bf2f((unsigned short)ua) * w);
        atomicAdd(ob + j, bf2f((unsigned short)ub) * w);
    }
}

// ---------------------------------------------------------------------------
// 5) embed = x2 + (A_on_x2 + b_on); predictor inputs -> bf16.
// ---------------------------------------------------------------------------
__global__ void __launch_bounds__(256)
embed_prep(const float* __restrict__ x, const float* __restrict__ perb,
           const float* __restrict__ Aon_x, const float* __restrict__ Aon_x2,
           const float* __restrict__ b_on,
           float* __restrict__ embed,
           unsigned short* __restrict__ onxb, unsigned short* __restrict__ onyb, int nd)
{
    int i = blockIdx.x * blockDim.x + threadIdx.x;
    if (i >= nd) return;
    int   c  = i & 127;
    float b  = b_on[c];
    float ox = Aon_x[i]  + b;   // online_x = gcn(x, W_on)
    float oy = Aon_x2[i] + b;   // online_y = gcn(x2, W_on)
    embed[i] = x[i] + perb[i] + oy;
    onxb[i]  = f2bf(ox);
    onyb[i]  = f2bf(oy);
}

// ---------------------------------------------------------------------------
// 6a) BatchNorm stats: per-column sum/sumsq via LDS float atomics.
// ---------------------------------------------------------------------------
__global__ void __launch_bounds__(256)
bn_stats(const unsigned short* __restrict__ H, float* __restrict__ stats, int nd)
{
    __shared__ float ls[256];
    int tid = threadIdx.x;
    ls[tid] = 0.f;
    __syncthreads();
    for (int i = blockIdx.x * 256 + tid; i < nd; i += gridDim.x * 256) {
        float v = bf2f(H[i]);
        int   c = i & 127;
        atomicAdd(&ls[c],       v);
        atomicAdd(&ls[128 + c], v * v);
    }
    __syncthreads();
    atomicAdd(&stats[tid], ls[tid]);
}

// ---------------------------------------------------------------------------
// 6b) BN normalize + affine + PReLU -> bf16.
// ---------------------------------------------------------------------------
__global__ void __launch_bounds__(256)
bn_apply(const unsigned short* __restrict__ H, const float* __restrict__ stats,
         const float* __restrict__ g, const float* __restrict__ beta,
         const float* __restrict__ pa, unsigned short* __restrict__ outb, int nd)
{
    int i = blockIdx.x * blockDim.x + threadIdx.x;
    if (i >= nd) return;
    int   c   = i & 127;
    float inv = 1.0f / (float)NROW;
    float mu  = stats[c] * inv;
    float var = stats[128 + c] * inv - mu * mu;
    float v   = bf2f(H[i]);
    float h   = (v - mu) * rsqrtf(var + 1e-5f) * g[c] + beta[c];
    float a   = pa[0];
    h = h > 0.f ? h : a * h;
    outb[i] = f2bf(h);
}

// ---------------------------------------------------------------------------
// 7) Cosine loss: wave per row, shfl_xor wave32 reductions, atomic accumulate.
//    loss += (2 - 2*cos(px,tx)) + (2 - 2*cos(py,ty)), averaged over rows.
// ---------------------------------------------------------------------------
__global__ void __launch_bounds__(256)
loss_kernel(const float* __restrict__ px, const float* __restrict__ py,
            const float* __restrict__ tx, const float* __restrict__ ty,
            const float* __restrict__ btg, float* __restrict__ lossOut, int nrows)
{
    int wave = threadIdx.x >> 5, lane = threadIdx.x & 31;
    int row = blockIdx.x * 8 + wave;
    if (row >= nrows) return;
    int co = lane * 4;
    float4 p1 = *(const float4*)(px  + (size_t)row * DN + co);
    float4 t1 = *(const float4*)(tx  + (size_t)row * DN + co);
    float4 p2 = *(const float4*)(py  + (size_t)row * DN + co);
    float4 t2 = *(const float4*)(ty  + (size_t)row * DN + co);
    float4 bb = *(const float4*)(btg + co);
    const float* P1 = (const float*)&p1; const float* T1 = (const float*)&t1;
    const float* P2 = (const float*)&p2; const float* T2 = (const float*)&t2;
    const float* B  = (const float*)&bb;
    float d1 = 0.f, n1p = 0.f, n1t = 0.f, d2 = 0.f, n2p = 0.f, n2t = 0.f;
#pragma unroll
    for (int j = 0; j < 4; ++j) {
        float a1 = P1[j], b1 = T1[j] + B[j];
        float a2 = P2[j], b2 = T2[j] + B[j];
        d1 += a1 * b1; n1p += a1 * a1; n1t += b1 * b1;
        d2 += a2 * b2; n2p += a2 * a2; n2t += b2 * b2;
    }
#pragma unroll
    for (int off = 16; off > 0; off >>= 1) {
        d1  += __shfl_xor(d1,  off, 32);
        n1p += __shfl_xor(n1p, off, 32);
        n1t += __shfl_xor(n1t, off, 32);
        d2  += __shfl_xor(d2,  off, 32);
        n2p += __shfl_xor(n2p, off, 32);
        n2t += __shfl_xor(n2t, off, 32);
    }
    if (lane == 0) {
        const float eps = 1e-12f;
        float c1 = d1 / (fmaxf(sqrtf(n1p), eps) * fmaxf(sqrtf(n1t), eps));
        float c2 = d2 / (fmaxf(sqrtf(n2p), eps) * fmaxf(sqrtf(n2t), eps));
        atomicAdd(lossOut, (4.0f - 2.0f * (c1 + c2)) * (1.0f / (float)NROW));
    }
}

// ---------------------------------------------------------------------------
// Host-side launch sequence.
// ---------------------------------------------------------------------------
extern "C" void kernel_launch(void* const* d_in, const int* in_sizes, int n_in,
                              void* d_out, int out_size, void* d_ws, size_t ws_size,
                              hipStream_t stream) {
    (void)in_sizes; (void)n_in; (void)out_size; (void)ws_size;

    const float* x    = (const float*)d_in[0];
    const float* perb = (const float*)d_in[1];
    const int*   er   = (const int*)d_in[2];
    const int*   ec   = (const int*)d_in[3];
    const float* ew   = (const float*)d_in[4];
    const float* W_on = (const float*)d_in[5];
    const float* b_on = (const float*)d_in[6];
    const float* W_tg = (const float*)d_in[7];
    const float* b_tg = (const float*)d_in[8];
    const float* W1   = (const float*)d_in[9];
    const float* b1   = (const float*)d_in[10];
    const float* bn_g = (const float*)d_in[11];
    const float* bn_b = (const float*)d_in[12];
    const float* p_a  = (const float*)d_in[13];
    const float* W2   = (const float*)d_in[14];
    const float* b2   = (const float*)d_in[15];

    const size_t ND = (size_t)NROW * DN;  // 6,400,000

    // Workspace carve (with lifetime-based reuse; ~180 MB peak).
    char* w = (char*)d_ws;
    unsigned short* xb   = (unsigned short*)w; w += ND * 2;
    unsigned short* x2b  = (unsigned short*)w; w += ND * 2;
    unsigned short* Hb0  = (unsigned short*)w; w += ND * 2;  // H_on(x)
    unsigned short* Hb1  = (unsigned short*)w; w += ND * 2;  // H_on(x2)
    unsigned short* Hb2  = (unsigned short*)w; w += ND * 2;  // H_tg(x)
    unsigned short* Hb3  = (unsigned short*)w; w += ND * 2;  // H_tg(x2)
    float* Aon_x  = (float*)w; w += ND * 4;
    float* Aon_x2 = (float*)w; w += ND * 4;
    float* Atg_x  = (float*)w; w += ND * 4;
    float* Atg_x2 = (float*)w; w += ND * 4;
    unsigned short* WTon = (unsigned short*)w; w += DN * DN * 2;
    unsigned short* WTtg = (unsigned short*)w; w += DN * DN * 2;
    unsigned short* WT1  = (unsigned short*)w; w += DN * DN * 2;
    unsigned short* WT2  = (unsigned short*)w; w += DN * DN * 2;
    float* stats = (float*)w; w += 512 * 4;   // [x: sum|sumsq][y: sum|sumsq]

    // Reused regions (all dead at point of reuse; see pipeline order):
    unsigned short* onxb = Hb0;            // predictor input x (after agg)
    unsigned short* onyb = Hb1;            // predictor input y
    unsigned short* H1x  = Hb2;            // predictor hidden x (after agg)
    unsigned short* H1y  = Hb3;            // predictor hidden y
    unsigned short* actx = xb;             // PReLU output x (after stage GEMMs)
    unsigned short* acty = x2b;
    float* px = (float*)Hb0;               // spans Hb0+Hb1 (after GEMM1)
    float* py = (float*)Hb2;               // spans Hb2+Hb3 (after bn_apply)

    float* embed   = (float*)d_out;
    float* lossOut = (float*)d_out + ND;

    const int tpb   = 256;
    const int gElem = (int)((ND + tpb - 1) / tpb);          // 25000
    const int gGemm = (NROW / 16 + 7) / 8;                  // 391
    const int gAgg  = (NEDGE * 32 + tpb - 1) / tpb;         // 100000
    const int gLoss = (NROW + 7) / 8;                       // 6250

    prep_main<<<gElem, tpb, 0, stream>>>(x, perb, xb, x2b,
                                         Aon_x, Aon_x2, Atg_x, Atg_x2,
                                         stats, lossOut, (int)ND);
    prep_w<<<(DN * DN + tpb - 1) / tpb, tpb, 0, stream>>>(W_on, W_tg, W1, W2,
                                                          WTon, WTtg, WT1, WT2);

    // Stage GEMMs: H = X @ W (bf16 out, no bias — bias folded after agg)
    gemm_wmma<true, false><<<gGemm, tpb, 0, stream>>>(xb,  WTon, nullptr, Hb0, nullptr, NROW);
    gemm_wmma<true, false><<<gGemm, tpb, 0, stream>>>(x2b, WTon, nullptr, Hb1, nullptr, NROW);
    gemm_wmma<true, false><<<gGemm, tpb, 0, stream>>>(xb,  WTtg, nullptr, Hb2, nullptr, NROW);
    gemm_wmma<true, false><<<gGemm, tpb, 0, stream>>>(x2b, WTtg, nullptr, Hb3, nullptr, NROW);

    // Edge scatter-add (fused on/tg pairs sharing index loads)
    agg2_kernel<<<gAgg, tpb, 0, stream>>>(Hb0, Hb2, er, ec, ew, Aon_x,  Atg_x,  NEDGE);
    agg2_kernel<<<gAgg, tpb, 0, stream>>>(Hb1, Hb3, er, ec, ew, Aon_x2, Atg_x2, NEDGE);

    // embed + predictor inputs
    embed_prep<<<gElem, tpb, 0, stream>>>(x, perb, Aon_x, Aon_x2, b_on,
                                          embed, onxb, onyb, (int)ND);

    // Predictor: GEMM1 (+b1) -> BN stats -> BN+PReLU -> GEMM2 (+b2)
    gemm_wmma<true, true><<<gGemm, tpb, 0, stream>>>(onxb, WT1, b1, H1x, nullptr, NROW);
    gemm_wmma<true, true><<<gGemm, tpb, 0, stream>>>(onyb, WT1, b1, H1y, nullptr, NROW);
    bn_stats<<<1024, tpb, 0, stream>>>(H1x, stats,       (int)ND);
    bn_stats<<<1024, tpb, 0, stream>>>(H1y, stats + 256, (int)ND);
    bn_apply<<<gElem, tpb, 0, stream>>>(H1x, stats,       bn_g, bn_b, p_a, actx, (int)ND);
    bn_apply<<<gElem, tpb, 0, stream>>>(H1y, stats + 256, bn_g, bn_b, p_a, acty, (int)ND);
    gemm_wmma<false, true><<<gGemm, tpb, 0, stream>>>(actx, WT2, b2, nullptr, px, NROW);
    gemm_wmma<false, true><<<gGemm, tpb, 0, stream>>>(acty, WT2, b2, nullptr, py, NROW);

    // Loss: pair (px, tgt_x = Atg_x2+b_tg) and (py, tgt_y = Atg_x+b_tg)
    loss_kernel<<<gLoss, tpb, 0, stream>>>(px, py, Atg_x2, Atg_x, b_tg,
                                           lossOut, NROW);
}
